// GINLayer_7541962572001
// MI455X (gfx1250) — compile-verified, matrix operand order checked by reading, not run on hardware
//
#include <hip/hip_runtime.h>

typedef _Float16 v16h __attribute__((ext_vector_type(16)));
typedef _Float16 v8h  __attribute__((ext_vector_type(8)));
typedef _Float16 v4h  __attribute__((ext_vector_type(4)));
typedef float    v8f  __attribute__((ext_vector_type(8)));

#define N_NODES 100000
#define E_EDGES 640000
#define BN_EPS  1e-5f

// ---------------------------------------------------------------------------
// K0: transpose + convert W1/W2 (f32 [k][n]) -> f16 [n][k]; zero BN stats.
// ---------------------------------------------------------------------------
__global__ void prep_weights(const float* __restrict__ W1, const float* __restrict__ W2,
                             _Float16* __restrict__ WT1, _Float16* __restrict__ WT2,
                             float* __restrict__ gsum, float* __restrict__ gsq) {
  int tid = blockIdx.x * blockDim.x + threadIdx.x;
  if (tid < 128) { gsum[tid] = 0.0f; gsq[tid] = 0.0f; }
  for (int i = tid; i < 128 * 128; i += gridDim.x * blockDim.x) {
    int n = i >> 7, k = i & 127;
    WT1[n * 128 + k] = (_Float16)W1[k * 128 + n];
    WT2[n * 128 + k] = (_Float16)W2[k * 128 + n];
  }
}

// ---------------------------------------------------------------------------
// K1: hpre = (1 + eps) * x       (12.8M floats, float4 vectorized)
// ---------------------------------------------------------------------------
__global__ void init_hpre(const float* __restrict__ x, const float* __restrict__ epsp,
                          float* __restrict__ hpre) {
  long long i = (long long)blockIdx.x * blockDim.x + threadIdx.x;  // float4 index
  if (i >= (long long)N_NODES * 32) return;
  float s = 1.0f + *epsp;
  float4 v = ((const float4*)x)[i];
  v.x *= s; v.y *= s; v.z *= s; v.w *= s;
  ((float4*)hpre)[i] = v;
}

// ---------------------------------------------------------------------------
// K2: scatter-add: hpre[dst] += x[src].  One wave32 per edge; 4 floats/lane.
// ---------------------------------------------------------------------------
__global__ void scatter_edges(const float* __restrict__ x, const int* __restrict__ ei,
                              float* __restrict__ hpre) {
  long long tid = (long long)blockIdx.x * blockDim.x + threadIdx.x;
  int e = (int)(tid >> 5);
  if (e >= E_EDGES) return;
  int f = ((int)tid & 31) * 4;
  int src = ei[e];
  int dst = ei[E_EDGES + e];
  float4 v = *(const float4*)(x + (size_t)src * 128 + f);
  float* p = hpre + (size_t)dst * 128 + f;
  unsafeAtomicAdd(p + 0, v.x);
  unsafeAtomicAdd(p + 1, v.y);
  unsafeAtomicAdd(p + 2, v.z);
  unsafeAtomicAdd(p + 3, v.w);
}

// ---------------------------------------------------------------------------
// GEMM: Cout[m][n] = act(Ain)[m][:] . WT[n][:] + bias[n]
//   mode 0: act = identity; also accumulate per-channel sum / sumsq (BN stats)
//   mode 1: act = relu(v*scale[k] + shift[k])   (BN applied while staging A)
// Block: 64 rows x 128 cols, 256 threads (8 waves); A-panel staged in LDS f16.
// Bounds check is block-uniform: only the last of 1563 blocks takes the
// guarded path, so the hot path has branch-free stores + stats.
// ---------------------------------------------------------------------------
__global__ void __launch_bounds__(256)
gin_gemm(const float* __restrict__ Ain, const _Float16* __restrict__ WT,
         const float* __restrict__ bias, float* __restrict__ Cout,
         float* __restrict__ gsum, float* __restrict__ gsq,
         const float* __restrict__ scale, const float* __restrict__ shift,
         int mode) {
  __shared__ _Float16 As[64 * 128];   // 16 KB staged A panel (f16)
  __shared__ float sA[128];           // mode0: sum accum | mode1: scale
  __shared__ float sB[128];           // mode0: sumsq accum | mode1: shift

  const int  tid  = threadIdx.x;
  const int  m0   = blockIdx.x * 64;
  const bool full = (m0 + 64 <= N_NODES);   // block-uniform

  if (mode == 0) { if (tid < 128) { sA[tid] = 0.0f; sB[tid] = 0.0f; } }
  else           { if (tid < 128) { sA[tid] = scale[tid]; sB[tid] = shift[tid]; } }
  __syncthreads();

  // ---- stage 64x128 A panel into LDS as f16 (with BN+ReLU in mode 1) ----
#pragma unroll
  for (int i = 0; i < 8; ++i) {
    int q  = tid + i * 256;        // float4 index in panel (2048 total)
    int r  = q >> 5;               // local row
    int c4 = (q & 31) * 4;         // column
    int m  = m0 + r;
    float4 v;
    if (full || m < N_NODES) v = *(const float4*)(Ain + (size_t)m * 128 + c4);
    else                     v = make_float4(0.f, 0.f, 0.f, 0.f);
    if (mode == 1) {
      v.x = fmaxf(fmaf(v.x, sA[c4 + 0], sB[c4 + 0]), 0.0f);
      v.y = fmaxf(fmaf(v.y, sA[c4 + 1], sB[c4 + 1]), 0.0f);
      v.z = fmaxf(fmaf(v.z, sA[c4 + 2], sB[c4 + 2]), 0.0f);
      v.w = fmaxf(fmaf(v.w, sA[c4 + 3], sB[c4 + 3]), 0.0f);
    }
    v4h h;
    h[0] = (_Float16)v.x; h[1] = (_Float16)v.y;
    h[2] = (_Float16)v.z; h[3] = (_Float16)v.w;
    *(v4h*)(&As[r * 128 + c4]) = h;
  }
  __syncthreads();

  const int wave = tid >> 5;
  const int lane = tid & 31;
  const int n    = lane & 15;          // A row in tile / B-D column
  const int kg   = (lane >> 4) * 8;    // K sub-chunk base within 32-chunk
  const int col  = wave * 16 + n;

  // ---- load B fragments: WT[n][k] contiguous along k (f16, 16B chunks) ----
  v16h Bt[4];
  const _Float16* wbase = WT + (size_t)col * 128;
#pragma unroll
  for (int ks = 0; ks < 4; ++ks) {
    v8h lo = *(const v8h*)(wbase + ks * 32 + kg);
    v8h hi = *(const v8h*)(wbase + ks * 32 + kg + 16);
#pragma unroll
    for (int j = 0; j < 8; ++j) { Bt[ks][j] = lo[j]; Bt[ks][8 + j] = hi[j]; }
  }

  const float bn = bias[col];
  float s = 0.0f, qacc = 0.0f;

#pragma unroll
  for (int t = 0; t < 4; ++t) {
    v8f c = {};
    const _Float16* abase = &As[(t * 16 + n) * 128 + kg];
#pragma unroll
    for (int ks = 0; ks < 4; ++ks) {
      v8h lo = *(const v8h*)(abase + ks * 32);
      v8h hi = *(const v8h*)(abase + ks * 32 + 16);
      v16h a;
#pragma unroll
      for (int j = 0; j < 8; ++j) { a[j] = lo[j]; a[8 + j] = hi[j]; }
      c = __builtin_amdgcn_wmma_f32_16x16x32_f16(false, a, false, Bt[ks],
                                                 (short)0, c, false, false);
    }
    const int mbase = m0 + t * 16 + (lane >> 4) * 8;
    float* outp = Cout + (size_t)mbase * 128 + col;
    if (full) {
      // hot path: no per-element branches, no exec-mask churn
#pragma unroll
      for (int r = 0; r < 8; ++r) {
        float v = c[r] + bn;
        outp[(size_t)r * 128] = v;
        s += v; qacc += v * v;
      }
    } else {
#pragma unroll
      for (int r = 0; r < 8; ++r) {
        if (mbase + r < N_NODES) {
          float v = c[r] + bn;
          outp[(size_t)r * 128] = v;
          s += v; qacc += v * v;
        }
      }
    }
  }

  if (mode == 0) {
    atomicAdd(&sA[col], s);         // ds_add_f32; lanes L and L+16 share col
    atomicAdd(&sB[col], qacc);
    __syncthreads();
    if (tid < 128) {
      unsafeAtomicAdd(&gsum[tid], sA[tid]);
      unsafeAtomicAdd(&gsq[tid],  sB[tid]);
    }
  }
}

// ---------------------------------------------------------------------------
// K5: BN finalize -> per-channel scale/shift
// ---------------------------------------------------------------------------
__global__ void bn_finalize(const float* __restrict__ gsum, const float* __restrict__ gsq,
                            const float* __restrict__ gamma, const float* __restrict__ beta,
                            float* __restrict__ scale, float* __restrict__ shift) {
  int c = threadIdx.x;
  if (c >= 128) return;
  float inv_n = 1.0f / (float)N_NODES;
  float mu  = gsum[c] * inv_n;
  float var = gsq[c] * inv_n - mu * mu;
  float r   = rsqrtf(var + BN_EPS);
  float sc  = gamma[c] * r;
  scale[c] = sc;
  shift[c] = beta[c] - mu * sc;
}

// ---------------------------------------------------------------------------
extern "C" void kernel_launch(void* const* d_in, const int* in_sizes, int n_in,
                              void* d_out, int out_size, void* d_ws, size_t ws_size,
                              hipStream_t stream) {
  const float* x     = (const float*)d_in[0];
  const int*   ei    = (const int*)d_in[1];
  const float* W1    = (const float*)d_in[2];
  const float* b1    = (const float*)d_in[3];
  const float* gamma = (const float*)d_in[4];
  const float* beta  = (const float*)d_in[5];
  const float* W2    = (const float*)d_in[6];
  const float* b2    = (const float*)d_in[7];
  const float* eps   = (const float*)d_in[8];
  float* out = (float*)d_out;

  char* ws = (char*)d_ws;
  const size_t NODE_BYTES = (size_t)N_NODES * 128 * sizeof(float);  // 51.2 MB
  float*    hpre = (float*)ws;
  float*    h1   = (float*)(ws + NODE_BYTES);
  _Float16* WT1  = (_Float16*)(ws + 2 * NODE_BYTES);
  _Float16* WT2  = (_Float16*)(ws + 2 * NODE_BYTES + 32768);
  float*    gsum = (float*)(ws + 2 * NODE_BYTES + 65536);
  float*    gsq  = gsum + 128;
  float*    scl  = gsum + 256;
  float*    shf  = gsum + 384;

  prep_weights<<<64, 256, 0, stream>>>(W1, W2, WT1, WT2, gsum, gsq);

  init_hpre<<<(N_NODES * 32 + 255) / 256, 256, 0, stream>>>(x, eps, hpre);

  scatter_edges<<<(E_EDGES * 32) / 256, 256, 0, stream>>>(x, ei, hpre);

  const int gblocks = (N_NODES + 63) / 64;  // 1563
  gin_gemm<<<gblocks, 256, 0, stream>>>(hpre, WT1, b1, h1, gsum, gsq,
                                        nullptr, nullptr, 0);

  bn_finalize<<<1, 128, 0, stream>>>(gsum, gsq, gamma, beta, scl, shf);

  gin_gemm<<<gblocks, 256, 0, stream>>>(h1, WT2, b2, out, nullptr, nullptr,
                                        scl, shf, 1);
}
